// GlobalFeat_40183714021761
// MI455X (gfx1250) — matrix operand order repaired and checked by reading, not compiled, not run on hardware
//
#include <hip/hip_runtime.h>

#define BATCH 4
#define NPTS  2048
#define KNBR  30
#define EPSV  1e-5f
#define SLOPE 0.2f
#define LDCAT 512
#define GEMM_WAVES 4
#define STATS_WG 240

typedef __attribute__((ext_vector_type(16))) __bf16 v16bf;
typedef __attribute__((ext_vector_type(8)))  float  v8f;

__device__ __forceinline__ unsigned short f2bf(float f) {
  unsigned int u = __float_as_uint(f);
  u += 0x7FFFu + ((u >> 16) & 1u);           // round-to-nearest-even
  return (unsigned short)(u >> 16);
}

__device__ __forceinline__ void load8bf(v16bf& d, int at, const unsigned short* p) {
  union { uint4 q; __bf16 h[8]; } t;
  t.q = *reinterpret_cast<const uint4*>(p);
#pragma unroll
  for (int i = 0; i < 8; ++i) d[at + i] = t.h[i];
}

#define WMMA_BF16(A_, B_, C_) \
  __builtin_amdgcn_wmma_f32_16x16x32_bf16(false, (A_), false, (B_), (short)0, (C_), false, false)

// ---------------------------------------------------------------------------
// Generic bf16 GEMM via v_wmma_f32_16x16x32_bf16, register-blocked 1x4 and
// software-pipelined: loads for K-step k+32 are issued before the 4 WMMAs of
// step k, so VMEM latency overlaps the matrix pipe (partial loadcnt waits).
//   A:  M x lda row-major bf16      (per-batch stride sA elements)
//   Bm: "B^T": Nn x ldb row-major   (per-batch stride sB)
//   C:  C[n*ldc + m] f32            (per-batch stride sC)
// Requirements: M mult 16, Kk mult 32, Nn mult 256 (= GEMM_WAVES * 64).
// ---------------------------------------------------------------------------
__global__ void wmma_gemm(const unsigned short* __restrict__ A,
                          const unsigned short* __restrict__ Bm,
                          float* __restrict__ C,
                          int M, int Nn, int Kk, int lda, int ldb, int ldc,
                          long long sA, long long sB, long long sC) {
  const int lane = threadIdx.x & 31;
  const int wid  = threadIdx.x >> 5;
  const int nT   = (blockIdx.x * GEMM_WAVES + wid) << 6;   // 64 columns per wave
  const int mT   = blockIdx.y << 4;
  const int bb   = blockIdx.z;
  const int hi   = lane >> 4;
  const int l15  = lane & 15;

  const unsigned short* Ab = A  + sA * bb + (long long)(mT + l15) * lda + hi * 16;
  const unsigned short* Bb = Bm + sB * bb + (long long)(nT + l15) * ldb + hi * 16;
  const long long bstep = (long long)16 * ldb;

  // Prologue: fragments for K-step 0.
  v16bf av, b0, b1, b2, b3;
  load8bf(av, 0, Ab);            load8bf(av, 8, Ab + 8);
  load8bf(b0, 0, Bb);            load8bf(b0, 8, Bb + 8);
  load8bf(b1, 0, Bb + bstep);    load8bf(b1, 8, Bb + bstep + 8);
  load8bf(b2, 0, Bb + 2*bstep);  load8bf(b2, 8, Bb + 2*bstep + 8);
  load8bf(b3, 0, Bb + 3*bstep);  load8bf(b3, 8, Bb + 3*bstep + 8);

  v8f acc0 = {}, acc1 = {}, acc2 = {}, acc3 = {};
  for (int k0 = 32; k0 < Kk; k0 += 32) {
    // Issue next K-step's loads first (independent of current WMMAs).
    v16bf na, n0, n1, n2, n3;
    load8bf(na, 0, Ab + k0);            load8bf(na, 8, Ab + k0 + 8);
    load8bf(n0, 0, Bb + k0);            load8bf(n0, 8, Bb + k0 + 8);
    load8bf(n1, 0, Bb + bstep + k0);    load8bf(n1, 8, Bb + bstep + k0 + 8);
    load8bf(n2, 0, Bb + 2*bstep + k0);  load8bf(n2, 8, Bb + 2*bstep + k0 + 8);
    load8bf(n3, 0, Bb + 3*bstep + k0);  load8bf(n3, 8, Bb + 3*bstep + k0 + 8);
    acc0 = WMMA_BF16(av, b0, acc0);
    acc1 = WMMA_BF16(av, b1, acc1);
    acc2 = WMMA_BF16(av, b2, acc2);
    acc3 = WMMA_BF16(av, b3, acc3);
    av = na; b0 = n0; b1 = n1; b2 = n2; b3 = n3;
  }
  acc0 = WMMA_BF16(av, b0, acc0);
  acc1 = WMMA_BF16(av, b1, acc1);
  acc2 = WMMA_BF16(av, b2, acc2);
  acc3 = WMMA_BF16(av, b3, acc3);

  // D layout: lane holds column n = l15 (+16*t), rows m = mT + r + 8*hi
  float* Cp = C + sC * bb + (long long)(nT + l15) * ldc + mT + hi * 8;
  const long long cstep = (long long)16 * ldc;
#pragma unroll
  for (int r = 0; r < 8; ++r) Cp[r] = acc0[r];
#pragma unroll
  for (int r = 0; r < 8; ++r) Cp[cstep + r] = acc1[r];
#pragma unroll
  for (int r = 0; r < 8; ++r) Cp[2 * cstep + r] = acc2[r];
#pragma unroll
  for (int r = 0; r < 8; ++r) Cp[3 * cstep + r] = acc3[r];
}

// ---------------------------------------------------------------------------
// Packing kernels (f32 -> bf16, zero-padded channels)
// ---------------------------------------------------------------------------
__global__ void pack_x0(const float* __restrict__ x, unsigned short* __restrict__ x0T) {
  int i = blockIdx.x * blockDim.x + threadIdx.x;        // B*N*32
  if (i >= BATCH * NPTS * 32) return;
  int c = i & 31, n = (i >> 5) % NPTS, b = i / (32 * NPTS);
  float v = (c < 3) ? x[((long long)b * 3 + c) * NPTS + n] : 0.f;
  x0T[i] = f2bf(v);
}

__global__ void pack_w_edge(const float* __restrict__ W, unsigned short* __restrict__ w1,
                            unsigned short* __restrict__ w2, int O, int C, int Cpad) {
  int i = blockIdx.x * blockDim.x + threadIdx.x;
  if (i >= O * Cpad) return;
  int c = i % Cpad, o = i / Cpad;
  float a = 0.f, d = 0.f;
  if (c < C) { a = W[(long long)o * 2 * C + c]; d = W[(long long)o * 2 * C + C + c] - a; }
  w1[i] = f2bf(a);
  w2[i] = f2bf(d);
}

__global__ void pack_w5(const float* __restrict__ W, unsigned short* __restrict__ wb) {
  int i = blockIdx.x * blockDim.x + threadIdx.x;
  if (i >= 1024 * 512) return;
  wb[i] = f2bf(W[i]);
}

// ---------------------------------------------------------------------------
// Per-point squared norms. xf layout (B, C, N) f32.
// ---------------------------------------------------------------------------
__global__ void norms_k(const float* __restrict__ xf, float* __restrict__ xx, int C) {
  int i = blockIdx.x * blockDim.x + threadIdx.x;
  if (i >= BATCH * NPTS) return;
  int b = i / NPTS, m = i % NPTS;
  const float* p = xf + (long long)b * C * NPTS + m;
  float s = 0.f;
  for (int c = 0; c < C; ++c) { float t = p[(long long)c * NPTS]; s += t * t; }
  xx[i] = s;
}

// ---------------------------------------------------------------------------
// Top-k(30) over pd[n,m] = 2*G[n,m] - xx[n] - xx[m]; one block per row n.
// Tie-break: lowest index (matches lax.top_k stability).
// ---------------------------------------------------------------------------
__global__ void knn_topk(const float* __restrict__ G, const float* __restrict__ xxb,
                         int* __restrict__ idxOut) {
  __shared__ float pd[NPTS];
  __shared__ float bv[256];
  __shared__ int   bi[256];
  const int n = blockIdx.x, tid = threadIdx.x;
  const float xn = xxb[n];
  for (int t = tid; t < NPTS; t += 256)
    pd[t] = 2.0f * G[(long long)n * NPTS + t] - xn - xxb[t];
  __syncthreads();
  for (int kk = 0; kk < KNBR; ++kk) {
    float best = -3.0e38f; int besti = 0;
    for (int t = tid; t < NPTS; t += 256) {
      float v = pd[t];
      if (v > best) { best = v; besti = t; }
    }
    bv[tid] = best; bi[tid] = besti;
    __syncthreads();
    for (int s = 128; s > 0; s >>= 1) {
      if (tid < s) {
        if (bv[tid + s] > bv[tid] || (bv[tid + s] == bv[tid] && bi[tid + s] < bi[tid])) {
          bv[tid] = bv[tid + s]; bi[tid] = bi[tid + s];
        }
      }
      __syncthreads();
    }
    if (tid == 0) { idxOut[n * KNBR + kk] = bi[0]; pd[bi[0]] = -3.0e38f; }
    __syncthreads();
  }
}

// ---------------------------------------------------------------------------
// BN statistics, edge layers: y(b,o,n,kk) = u[b,idx,o] + v[b,n,o].
// Deterministic two-stage reduction. blockDim = O, one channel per thread.
// ---------------------------------------------------------------------------
__global__ void stats_edge(const float* __restrict__ u, const float* __restrict__ v,
                           const int* __restrict__ idx, float* __restrict__ part,
                           int O, int itemsPerWG) {
  const int o = threadIdx.x, wg = blockIdx.x;
  const long long start = (long long)wg * itemsPerWG;
  float s = 0.f, s2 = 0.f;
  for (int it = 0; it < itemsPerWG; ++it) {
    long long e  = start + it;
    long long bn = e / KNBR;
    int m = idx[e];
    int b = (int)(bn / NPTS);
    float val = u[((long long)b * NPTS + m) * O + o] + v[bn * O + o];
    s += val; s2 += val * val;
  }
  part[((long long)wg * 2)     * O + o] = s;
  part[((long long)wg * 2 + 1) * O + o] = s2;
}

__global__ void stats_plain(const float* __restrict__ u, float* __restrict__ part,
                            int O, int itemsPerWG) {
  const int o = threadIdx.x, wg = blockIdx.x;
  const long long start = (long long)wg * itemsPerWG;
  float s = 0.f, s2 = 0.f;
  for (int it = 0; it < itemsPerWG; ++it) {
    float val = u[(start + it) * O + o];
    s += val; s2 += val * val;
  }
  part[((long long)wg * 2)     * O + o] = s;
  part[((long long)wg * 2 + 1) * O + o] = s2;
}

__global__ void bn_finalize(const float* __restrict__ part, int nwg, int O, float invCount,
                            const float* __restrict__ g, const float* __restrict__ bias,
                            float* __restrict__ cs) {
  int o = blockIdx.x * blockDim.x + threadIdx.x;
  if (o >= O) return;
  float s = 0.f, s2 = 0.f;
  for (int w = 0; w < nwg; ++w) {
    s  += part[((long long)w * 2)     * O + o];
    s2 += part[((long long)w * 2 + 1) * O + o];
  }
  float mean  = s * invCount;
  float var   = s2 * invCount - mean * mean;
  float scale = g[o] * rsqrtf(var + EPSV);
  cs[2 * o]     = scale;
  cs[2 * o + 1] = bias[o] - mean * scale;
}

// ---------------------------------------------------------------------------
// Apply BN + leaky relu + max over neighbors. grid(N,B), blockDim=O.
// Writes next-layer f32 (B,O,N) for KNN and bf16 into catT (B,N,512) for GEMMs.
// ---------------------------------------------------------------------------
__global__ void apply_edge(const float* __restrict__ u, const float* __restrict__ v,
                           const int* __restrict__ idx, const float* __restrict__ cs,
                           float* __restrict__ xf_out, unsigned short* __restrict__ catT,
                           int O, int coff) {
  const int o = threadIdx.x, n = blockIdx.x, b = blockIdx.y;
  const long long bn = (long long)b * NPTS + n;
  const float scale = cs[2 * o], shift = cs[2 * o + 1];
  const float vv = v[bn * O + o];
  float mx = -3.0e38f;
  for (int kk = 0; kk < KNBR; ++kk) {
    int m = idx[bn * KNBR + kk];
    float y = u[((long long)b * NPTS + m) * O + o] + vv;
    y = y * scale + shift;
    y = (y > 0.f) ? y : y * SLOPE;
    mx = fmaxf(mx, y);
  }
  xf_out[((long long)b * O + o) * NPTS + n] = mx;
  catT[bn * LDCAT + coff + o] = f2bf(mx);
}

__global__ void apply_final(const float* __restrict__ u5, const float* __restrict__ cs,
                            float* __restrict__ out) {
  const int o = threadIdx.x, b = blockIdx.x;
  const float scale = cs[2 * o], shift = cs[2 * o + 1];
  float mx = -3.0e38f;
  for (int n = 0; n < NPTS; ++n) {
    float y = u5[((long long)b * NPTS + n) * 1024 + o] * scale + shift;
    y = (y > 0.f) ? y : y * SLOPE;
    mx = fmaxf(mx, y);
  }
  out[b * 1024 + o] = mx;
}

// ---------------------------------------------------------------------------
extern "C" void kernel_launch(void* const* d_in, const int* in_sizes, int n_in,
                              void* d_out, int out_size, void* d_ws, size_t ws_size,
                              hipStream_t stream) {
  (void)in_sizes; (void)n_in; (void)out_size; (void)ws_size;
  const float* x = (const float*)d_in[0];
  const float *W[5], *gg[5], *bb_[5];
  for (int i = 0; i < 5; ++i) {
    W[i]   = (const float*)d_in[1 + 3 * i];
    gg[i]  = (const float*)d_in[2 + 3 * i];
    bb_[i] = (const float*)d_in[3 + 3 * i];
  }

  char* ws = (char*)d_ws;
  size_t off = 0;
  auto alloc = [&](size_t bytes) -> char* {
    char* p = ws + off;
    off = (off + bytes + 255) & ~(size_t)255;
    return p;
  };
  unsigned short* x0T  = (unsigned short*)alloc((size_t)BATCH * NPTS * 32 * 2);
  unsigned short* catT = (unsigned short*)alloc((size_t)BATCH * NPTS * 512 * 2);
  float* xf   = (float*)alloc((size_t)BATCH * 256 * NPTS * 4);
  float* xx   = (float*)alloc((size_t)BATCH * NPTS * 4);
  float* G    = (float*)alloc((size_t)NPTS * NPTS * 4);          // per-batch, reused
  int*   idxb = (int*)  alloc((size_t)BATCH * NPTS * KNBR * 4);
  unsigned short* w1b = (unsigned short*)alloc((size_t)256 * 128 * 2);
  unsigned short* w2b = (unsigned short*)alloc((size_t)256 * 128 * 2);
  unsigned short* w5b = (unsigned short*)alloc((size_t)1024 * 512 * 2);
  float* u    = (float*)alloc((size_t)BATCH * NPTS * 256 * 4);
  float* vb   = (float*)alloc((size_t)BATCH * NPTS * 256 * 4);
  float* u5   = (float*)alloc((size_t)BATCH * NPTS * 1024 * 4);
  float* part = (float*)alloc((size_t)2 * STATS_WG * 1024 * 4);
  float* cs   = (float*)alloc((size_t)2048 * 4);

  pack_x0<<<(BATCH * NPTS * 32 + 255) / 256, 256, 0, stream>>>(x, x0T);
  pack_w5<<<(1024 * 512 + 255) / 256, 256, 0, stream>>>(W[4], w5b);

  const int Cs[4]      = {3, 64, 64, 128};
  const int Cpads[4]   = {32, 64, 64, 128};
  const int Os[4]      = {64, 64, 128, 256};
  const int coffIn[4]  = {0, 0, 64, 128};
  const int coffOut[4] = {0, 64, 128, 256};
  const int Eedge = BATCH * NPTS * KNBR;
  const int GX = NPTS / (64 * GEMM_WAVES);   // N-strips: 2048 / 256 = 8 blocks

  for (int L = 0; L < 4; ++L) {
    const int O = Os[L], C = Cs[L], Cpad = Cpads[L];
    pack_w_edge<<<(O * Cpad + 255) / 256, 256, 0, stream>>>(W[L], w1b, w2b, O, C, Cpad);

    const float* xin = (L == 0) ? x : xf;
    norms_k<<<(BATCH * NPTS + 255) / 256, 256, 0, stream>>>(xin, xx, C);

    const unsigned short* xT = (L == 0) ? x0T : (catT + coffIn[L]);
    const int ldx = (L == 0) ? 32 : LDCAT;
    const int Kk  = Cpad;   // 32, 64, 64, 128 (zero padding contributes 0)

    // KNN: gram via WMMA (per batch, G reused), then top-30
    for (int b = 0; b < BATCH; ++b) {
      const unsigned short* xTb = xT + (long long)b * NPTS * ldx;
      dim3 g1(GX, NPTS / 16, 1);
      wmma_gemm<<<g1, 32 * GEMM_WAVES, 0, stream>>>(xTb, xTb, G, NPTS, NPTS, Kk,
                                                    ldx, ldx, NPTS, 0, 0, 0);
      knn_topk<<<NPTS, 256, 0, stream>>>(G, xx + b * NPTS, idxb + (long long)b * NPTS * KNBR);
    }

    // u = W1 * x, v = (W2 - W1) * x  -> stored (B, N, O)
    dim3 g2(GX, O / 16, BATCH);
    wmma_gemm<<<g2, 32 * GEMM_WAVES, 0, stream>>>(w1b, xT, u, O, NPTS, Kk, Cpad, ldx, O,
                                                  0, (long long)NPTS * ldx, (long long)NPTS * O);
    wmma_gemm<<<g2, 32 * GEMM_WAVES, 0, stream>>>(w2b, xT, vb, O, NPTS, Kk, Cpad, ldx, O,
                                                  0, (long long)NPTS * ldx, (long long)NPTS * O);

    stats_edge<<<STATS_WG, O, 0, stream>>>(u, vb, idxb, part, O, Eedge / STATS_WG);
    bn_finalize<<<(O + 255) / 256, 256, 0, stream>>>(part, STATS_WG, O, 1.0f / (float)Eedge,
                                                     gg[L], bb_[L], cs);
    apply_edge<<<dim3(NPTS, BATCH), O, 0, stream>>>(u, vb, idxb, cs, xf, catT, O, coffOut[L]);
  }

  // Final: y5 = W5(1024x512) * cat -> (B, N, 1024), BN over (B,N), lrelu, max over N
  dim3 g5(GX, 1024 / 16, BATCH);
  wmma_gemm<<<g5, 32 * GEMM_WAVES, 0, stream>>>(w5b, catT, u5, 1024, NPTS, 512, 512, LDCAT, 1024,
                                                0, (long long)NPTS * LDCAT, (long long)NPTS * 1024);
  stats_plain<<<64, 1024, 0, stream>>>(u5, part, 1024, (BATCH * NPTS) / 64);
  bn_finalize<<<4, 256, 0, stream>>>(part, 64, 1024, 1.0f / (float)(BATCH * NPTS),
                                     gg[4], bb_[4], cs);
  apply_final<<<BATCH, 1024, 0, stream>>>(u5, cs, (float*)d_out);
}
